// Q8Linear_74680891343684
// MI455X (gfx1250) — 8-bit matrix operand order repaired and checked by reading, compiled, not run on hardware
//
#include <hip/hip_runtime.h>
#include <stdint.h>


#define TOKENS 8192
#define IN_F   4096
#define OUT_F  16384

#define BM 128
#define BN 128
#define BK 64
#define KTILES (IN_F / BK)

typedef int          v8i __attribute__((ext_vector_type(8)));
typedef int          v4i __attribute__((ext_vector_type(4)));
typedef unsigned int v4u __attribute__((ext_vector_type(4)));

#if defined(__gfx1250__) && __has_builtin(__builtin_amdgcn_tensor_load_to_lds) && \
    __has_builtin(__builtin_amdgcn_s_wait_tensorcnt)
#define USE_TDM 1
#else
#define USE_TDM 0
#endif

#if !USE_TDM && defined(__gfx1250__) && __has_builtin(__builtin_amdgcn_global_load_async_to_lds_b128)
#define USE_ASYNC 1
#else
#define USE_ASYNC 0
#endif

#if USE_ASYNC
typedef __attribute__((address_space(1))) v4i* gptr_v4i;
typedef __attribute__((address_space(3))) v4i* lptr_v4i;
#endif

__global__ __launch_bounds__(256)
void q8_pack_kernel(const int* __restrict__ x, const int* __restrict__ w,
                    int8_t* __restrict__ xq, int8_t* __restrict__ wq) {
    const long long i  = (long long)blockIdx.x * blockDim.x + threadIdx.x;
    const long long nx = (long long)TOKENS * IN_F / 4;
    const long long nw = (long long)OUT_F  * IN_F / 4;
    if (i < nx) {
        int4 v = ((const int4*)x)[i];
        int p = (v.x & 0xFF) | ((v.y & 0xFF) << 8) | ((v.z & 0xFF) << 16) | (v.w << 24);
        ((int*)xq)[i] = p;
    } else if (i < nx + nw) {
        const long long j = i - nx;
        int4 v = ((const int4*)w)[j];
        int p = (v.x & 0xFF) | ((v.y & 0xFF) << 8) | ((v.z & 0xFF) << 16) | (v.w << 24);
        ((int*)wq)[j] = p;
    }
}

#if USE_TDM
__device__ __forceinline__ uint32_t lds_byte_addr(const void* p) {
    return (uint32_t)(uintptr_t)(__attribute__((address_space(3))) const void*)p;
}

__device__ __forceinline__ void tdm_load_tile(const int8_t* gtile, uint32_t lds_addr,
                                              uint32_t rows_total) {
    const unsigned long long ga = (unsigned long long)(uintptr_t)gtile;
    v4u g0;
    g0.x = 0x1u;
    g0.y = lds_addr;
    g0.z = (uint32_t)ga;
    g0.w = ((uint32_t)(ga >> 32) & 0x01FFFFFFu) | (2u << 30);
    v8i g1;
    g1[0] = 0;
    g1[1] = (int)((IN_F & 0xFFFF) << 16);
    g1[2] = (int)((IN_F >> 16) | ((rows_total & 0xFFFFu) << 16));
    g1[3] = (int)((rows_total >> 16) | ((uint32_t)BK << 16));
    g1[4] = BM;
    g1[5] = IN_F;
    g1[6] = 0;
    g1[7] = 0;
    v4i z = {0, 0, 0, 0};
#if defined(__clang_major__) && (__clang_major__ >= 23)
    v8i gz = {};
    __builtin_amdgcn_tensor_load_to_lds(g0, g1, z, z, gz, 0);
#else
    __builtin_amdgcn_tensor_load_to_lds(g0, g1, z, z, 0);
#endif
}
#endif

union AFrag { v8i v; unsigned long long q[4]; };
union BFrag { v8i v; uint4 o[2]; };

__global__ __launch_bounds__(256)
void q8_gemm_wmma_kernel(const int8_t* __restrict__ xq, const int8_t* __restrict__ wq,
                         const float* __restrict__ bias, const float* __restrict__ xsc,
                         const float* __restrict__ wsc, float* __restrict__ out) {
    __shared__ __align__(16) int8_t As[2][BM * BK];
    __shared__ __align__(16) int8_t Bs[2][BN * BK];

    const int tid  = threadIdx.x;
    const int lane = tid & 31;
    const int wave = tid >> 5;
    const int wm   = wave >> 1;
    const int wn   = wave & 1;
    const int bm   = blockIdx.y * BM;
    const int bn   = blockIdx.x * BN;

    const int am  = lane & 15;
    const int ak  = (lane >> 4) * 8;
    const int bnn = lane & 15;
    const int bk  = (lane >> 4) * 8;

    v8i acc[2][4] = {};

    auto compute_tile = [&](int buf) {
        AFrag a[2];
        #pragma unroll
        for (int mt = 0; mt < 2; ++mt) {
            const int8_t* p = &As[buf][(wm * 32 + mt * 16 + am) * BK + ak];
            a[mt].q[0] = *(const unsigned long long*)(p);
            a[mt].q[1] = *(const unsigned long long*)(p + 16);
            a[mt].q[2] = *(const unsigned long long*)(p + 32);
            a[mt].q[3] = *(const unsigned long long*)(p + 48);
        }
        BFrag b[4];
        #pragma unroll
        for (int nt = 0; nt < 4; ++nt) {
            const int8_t* p = &Bs[buf][(wn * 64 + nt * 16 + bnn) * BK + bk];
            ((uint2*)&b[nt].v)[0] = *(const uint2*)(p);
            ((uint2*)&b[nt].v)[1] = *(const uint2*)(p + 16);
            ((uint2*)&b[nt].v)[2] = *(const uint2*)(p + 32);
            ((uint2*)&b[nt].v)[3] = *(const uint2*)(p + 48);
        }
        #pragma unroll
        for (int mt = 0; mt < 2; ++mt)
            #pragma unroll
            for (int nt = 0; nt < 4; ++nt)
                acc[mt][nt] = __builtin_amdgcn_wmma_i32_16x16x64_iu8(
                     true, a[mt].v,  true, b[nt].v,
                    acc[mt][nt],  false,  false);
    };

#if USE_TDM
    const uint32_t ldsA[2] = { lds_byte_addr(&As[0][0]), lds_byte_addr(&As[1][0]) };
    const uint32_t ldsB[2] = { lds_byte_addr(&Bs[0][0]), lds_byte_addr(&Bs[1][0]) };
    auto issue_tile = [&](int buf, int kt) {
        const int kbyte = kt * BK;
        tdm_load_tile(xq + (long long)bm * IN_F + kbyte, ldsA[buf], TOKENS);
        tdm_load_tile(wq + (long long)bn * IN_F + kbyte, ldsB[buf], OUT_F);
    };
    if (wave == 0) issue_tile(0, 0);
    int buf = 0;
    for (int kt = 0; kt < KTILES; ++kt) {
        if (wave == 0) __builtin_amdgcn_s_wait_tensorcnt(0);
        __syncthreads();
        if (wave == 0 && kt + 1 < KTILES) issue_tile(buf ^ 1, kt + 1);
        compute_tile(buf);
        __syncthreads();
        buf ^= 1;
    }
#elif USE_ASYNC
    auto issue_tile = [&](int buf, int kt) {
        const int kbyte = kt * BK;
        #pragma unroll
        for (int s = 0; s < 2; ++s) {
            const int i   = tid + s * 256;
            const int row = i >> 2;
            const int col = (i & 3) * 16;
            __builtin_amdgcn_global_load_async_to_lds_b128(
                (gptr_v4i)(xq + (long long)(bm + row) * IN_F + kbyte + col),
                (lptr_v4i)(&As[buf][row * BK + col]), 0, 0);
            __builtin_amdgcn_global_load_async_to_lds_b128(
                (gptr_v4i)(wq + (long long)(bn + row) * IN_F + kbyte + col),
                (lptr_v4i)(&Bs[buf][row * BK + col]), 0, 0);
        }
    };
    issue_tile(0, 0);
    int buf = 0;
    for (int kt = 0; kt < KTILES; ++kt) {
        asm volatile("s_wait_asynccnt 0" ::: "memory");
        __syncthreads();
        if (kt + 1 < KTILES) issue_tile(buf ^ 1, kt + 1);
        compute_tile(buf);
        __syncthreads();
        buf ^= 1;
    }
#else
    for (int kt = 0; kt < KTILES; ++kt) {
        const int kbyte = kt * BK;
        uint4 ra[2], rb[2];
        int rows[2], cols[2];
        #pragma unroll
        for (int s = 0; s < 2; ++s) {
            const int i = tid + s * 256;
            rows[s] = i >> 2;
            cols[s] = (i & 3) * 16;
            ra[s] = *(const uint4*)(xq + (long long)(bm + rows[s]) * IN_F + kbyte + cols[s]);
            rb[s] = *(const uint4*)(wq + (long long)(bn + rows[s]) * IN_F + kbyte + cols[s]);
        }
        __syncthreads();
        #pragma unroll
        for (int s = 0; s < 2; ++s) {
            *(uint4*)(&As[0][rows[s] * BK + cols[s]]) = ra[s];
            *(uint4*)(&Bs[0][rows[s] * BK + cols[s]]) = rb[s];
        }
        __syncthreads();
        compute_tile(0);
        __syncthreads();
    }
#endif

    const int rowhalf = (lane >> 4) * 8;
    const int ncol    = lane & 15;
    #pragma unroll
    for (int nt = 0; nt < 4; ++nt) {
        const int col = bn + wn * 64 + nt * 16 + ncol;
        const float cs = wsc[col];
        const float cb = bias[col];
        #pragma unroll
        for (int mt = 0; mt < 2; ++mt) {
            const int rbase = bm + wm * 32 + mt * 16 + rowhalf;
            #pragma unroll
            for (int r = 0; r < 8; ++r) {
                const int row = rbase + r;
                const float v = (float)acc[mt][nt][r] * xsc[row] * cs + cb;
                out[(long long)row * OUT_F + col] = v;
            }
        }
    }
}

extern "C" void kernel_launch(void* const* d_in, const int* in_sizes, int n_in,
                              void* d_out, int out_size, void* d_ws, size_t ws_size,
                              hipStream_t stream) {
    (void)in_sizes; (void)n_in; (void)out_size; (void)ws_size;
    const int*   x    = (const int*)d_in[0];
    const int*   w    = (const int*)d_in[1];
    const float* bias = (const float*)d_in[2];
    const float* xsc  = (const float*)d_in[3];
    const float* wsc  = (const float*)d_in[4];
    float*       out  = (float*)d_out;

    int8_t* xq = (int8_t*)d_ws;
    int8_t* wq = xq + (size_t)TOKENS * IN_F;

    const long long packs = ((long long)TOKENS * IN_F + (long long)OUT_F * IN_F) / 4;
    const int pack_blocks = (int)((packs + 255) / 256);
    q8_pack_kernel<<<pack_blocks, 256, 0, stream>>>(x, w, xq, wq);

    dim3 grid(OUT_F / BN, TOKENS / BM);
    q8_gemm_wmma_kernel<<<grid, 256, 0, stream>>>(xq, wq, bias, xsc, wsc, out);
}
